// AttnDecoderRNN_8615704396043
// MI455X (gfx1250) — compile-verified, matrix operand order checked by reading, not run on hardware
//
#include <hip/hip_runtime.h>
#include <hip/hip_bf16.h>

// ---------------------------------------------------------------------------
// AttnDecoderRNN on gfx1250: bf16 WMMA GEMMs + fused elementwise kernels.
// Shapes fixed by the reference: B=256, T_OUT=36, S_IN=96, D=H=1024, L=2.
// Register blocking: each wave computes a (MT*16) x (NT*16) output tile,
// so the big GEMMs run 8 WMMAs per 12 b128 loads instead of 4 per 10.
// ---------------------------------------------------------------------------

typedef __attribute__((ext_vector_type(16))) __bf16 v16bf;
typedef __attribute__((ext_vector_type(8)))  __bf16 v8bf;
typedef __attribute__((ext_vector_type(8)))  float  v8f;

#define B_    256
#define TOUT_ 36
#define SIN_  96
#define D_    1024
#define H_    1024
#define L_    2

__device__ __forceinline__ __bf16 to_bf(float x) { return (__bf16)x; }

// ---------------- conversion / copy helpers ----------------
__global__ void k_f2bf(const float* __restrict__ src, __bf16* __restrict__ dst, int n) {
  int i = blockIdx.x * blockDim.x + threadIdx.x;
  if (i < n) dst[i] = to_bf(src[i]);
}

__global__ void k_copy_f32(const float* __restrict__ src, float* __restrict__ dst, int n) {
  int i = blockIdx.x * blockDim.x + threadIdx.x;
  if (i < n) dst[i] = src[i];
}

// Per-step: cat = [bf16(x_t), bf16(h1)], plus bf16 snapshots of h0/h1.
__global__ void k_pack_step(const float* __restrict__ target, int t,
                            const float* __restrict__ h,
                            __bf16* __restrict__ cat,
                            __bf16* __restrict__ h0b,
                            __bf16* __restrict__ h1b) {
  int idx = blockIdx.x * blockDim.x + threadIdx.x;   // B*H threads
  int b = idx >> 10, d = idx & 1023;
  float x  = target[((size_t)b * TOUT_ + t) * D_ + d];
  float h0 = h[idx];
  float h1 = h[(size_t)B_ * H_ + idx];
  cat[(size_t)b * 2048 + d]        = to_bf(x);
  cat[(size_t)b * 2048 + 1024 + d] = to_bf(h1);
  h0b[idx] = to_bf(h0);
  h1b[idx] = to_bf(h1);
}

// ---------------- bf16 WMMA GEMM: C[MxN] = A[MxK] * W[NxK]^T + bias ----------------
// One wave computes a (MT*16) x (NT*16) output tile. wave32; EXEC all-ones.
template <int MT, int NT, bool RELU, bool OUT_BF16>
__global__ void k_gemm_wmma(const __bf16* __restrict__ A,
                            const __bf16* __restrict__ W,
                            const float* __restrict__ bias,
                            float* __restrict__ Cf,
                            __bf16* __restrict__ Cb,
                            int M, int N, int K, int ldc) {
  const int wave = threadIdx.x >> 5;
  const int lane = threadIdx.x & 31;
  const int wavesPerBlock = blockDim.x >> 5;
  const int nStrips = N / (NT * 16);
  int gw = blockIdx.x * wavesPerBlock + wave;
  int mStrip = gw / nStrips;
  int nStrip = gw % nStrips;
  if (mStrip >= M / (MT * 16)) return;   // wave-uniform: EXEC stays all-ones

  const int m0 = mStrip * MT * 16;
  const int n0 = nStrip * NT * 16;
  const int l15 = lane & 15;
  const int kb  = (lane < 16) ? 0 : 8;   // 16-bit 16x32 fragment lane layout

  const __bf16* Arow[MT];
#pragma unroll
  for (int mi = 0; mi < MT; ++mi)
    Arow[mi] = A + (size_t)(m0 + mi * 16 + l15) * K + kb;
  const __bf16* Brow[NT];
#pragma unroll
  for (int ni = 0; ni < NT; ++ni)
    Brow[ni] = W + (size_t)(n0 + ni * 16 + l15) * K + kb;

  v8f acc[MT][NT] = {};
  for (int k = 0; k < K; k += 32) {
    v16bf afrag[MT];
#pragma unroll
    for (int mi = 0; mi < MT; ++mi) {
      v8bf alo = *(const v8bf*)(Arow[mi] + k);
      v8bf ahi = *(const v8bf*)(Arow[mi] + k + 16);
#pragma unroll
      for (int e = 0; e < 8; ++e) { afrag[mi][e] = alo[e]; afrag[mi][e + 8] = ahi[e]; }
    }
#pragma unroll
    for (int ni = 0; ni < NT; ++ni) {
      v8bf blo = *(const v8bf*)(Brow[ni] + k);
      v8bf bhi = *(const v8bf*)(Brow[ni] + k + 16);
      v16bf bfrag;
#pragma unroll
      for (int e = 0; e < 8; ++e) { bfrag[e] = blo[e]; bfrag[e + 8] = bhi[e]; }
#pragma unroll
      for (int mi = 0; mi < MT; ++mi)
        acc[mi][ni] = __builtin_amdgcn_wmma_f32_16x16x32_bf16(
            false, afrag[mi], false, bfrag, (short)0, acc[mi][ni], false, false);
    }
  }

  // C layout: VGPR v, lanes0-15 -> M=mbase+v ; lanes16-31 -> M=mbase+8+v
  const int mhalf = (lane >> 4) ? 8 : 0;
#pragma unroll
  for (int mi = 0; mi < MT; ++mi) {
    const int mbase = m0 + mi * 16 + mhalf;
#pragma unroll
    for (int ni = 0; ni < NT; ++ni) {
      int n = n0 + ni * 16 + l15;
      float bv = bias ? bias[n] : 0.0f;
#pragma unroll
      for (int v = 0; v < 8; ++v) {
        float x = acc[mi][ni][v] + bv;
        if (RELU) x = fmaxf(x, 0.0f);
        size_t o = (size_t)(mbase + v) * ldc + n;
        if (OUT_BF16) Cb[o] = to_bf(x);
        else          Cf[o] = x;
      }
    }
  }
}

// ---------------- softmax over 96, one wave per row ----------------
__global__ void k_softmax96(const float* __restrict__ logits,
                            float* __restrict__ aw,
                            float* __restrict__ attn_out, int t) {
  int wave = threadIdx.x >> 5, lane = threadIdx.x & 31;
  int row = blockIdx.x * (blockDim.x >> 5) + wave;
  if (row >= B_) return;
  const float* in = logits + (size_t)row * SIN_;
  float v0 = in[lane], v1 = in[lane + 32], v2 = in[lane + 64];
  float m = fmaxf(v0, fmaxf(v1, v2));
#pragma unroll
  for (int o = 16; o > 0; o >>= 1) m = fmaxf(m, __shfl_xor(m, o, 32));
  float e0 = __expf(v0 - m), e1 = __expf(v1 - m), e2 = __expf(v2 - m);
  float s = e0 + e1 + e2;
#pragma unroll
  for (int o = 16; o > 0; o >>= 1) s += __shfl_xor(s, o, 32);
  float inv = 1.0f / s;
  e0 *= inv; e1 *= inv; e2 *= inv;
  float* arow = aw + (size_t)row * SIN_;
  arow[lane] = e0; arow[lane + 32] = e1; arow[lane + 64] = e2;
  float* orow = attn_out + ((size_t)row * TOUT_ + t) * SIN_;
  orow[lane] = e0; orow[lane + 32] = e1; orow[lane + 64] = e2;
}

// ---------------- w_enc[b,d] = sum_s aw[b,s] * enc[b,s,d] -> bf16 into cat ----------------
__global__ void k_wenc(const float* __restrict__ aw,
                       const float* __restrict__ enc,
                       __bf16* __restrict__ cat) {
  int idx = blockIdx.x * blockDim.x + threadIdx.x;  // B*H
  int b = idx >> 10, d = idx & 1023;
  const float* arow = aw + (size_t)b * SIN_;
  const float* e = enc + (size_t)b * SIN_ * H_ + d;
  float acc = 0.0f;
#pragma unroll 4
  for (int s = 0; s < SIN_; ++s) acc += arow[s] * e[(size_t)s * H_];
  cat[(size_t)b * 2048 + 1024 + d] = to_bf(acc);
}

// ---------------- fused GRU gates ----------------
__device__ __forceinline__ float sigmoidf(float x) { return 1.0f / (1.0f + __expf(-x)); }

__global__ void k_gru_gate(const float* __restrict__ gi,
                           const float* __restrict__ gh,
                           float* __restrict__ hlayer,        // (B,H) f32, updated in place
                           __bf16* __restrict__ h_bf_out) {   // bf16 copy for next GEMM
  int idx = blockIdx.x * blockDim.x + threadIdx.x;  // B*H
  int b = idx >> 10, j = idx & 1023;
  const float* gib = gi + (size_t)b * 3072;
  const float* ghb = gh + (size_t)b * 3072;
  float r = sigmoidf(gib[j] + ghb[j]);
  float z = sigmoidf(gib[1024 + j] + ghb[1024 + j]);
  float n = tanhf(gib[2048 + j] + r * ghb[2048 + j]);
  float ho = hlayer[idx];
  float hn = (1.0f - z) * n + z * ho;
  hlayer[idx] = hn;
  h_bf_out[idx] = to_bf(hn);
}

// ---------------- host-side GEMM launcher ----------------
template <int MT, int NT, bool RELU, bool OUT_BF16>
static void launch_gemm(hipStream_t s, const __bf16* A, const __bf16* W,
                        const float* bias, float* Cf, __bf16* Cb,
                        int M, int N, int K, int ldc) {
  int nStrips = N / (NT * 16);
  int waves = (M / (MT * 16)) * nStrips;
  int blocks = (waves + 7) / 8;
  k_gemm_wmma<MT, NT, RELU, OUT_BF16><<<blocks, 256, 0, s>>>(A, W, bias, Cf, Cb, M, N, K, ldc);
}

extern "C" void kernel_launch(void* const* d_in, const int* in_sizes, int n_in,
                              void* d_out, int out_size, void* d_ws, size_t ws_size,
                              hipStream_t stream) {
  const float* target = (const float*)d_in[0];
  const float* hidden = (const float*)d_in[1];
  const float* enc    = (const float*)d_in[2];
  const float* attnW  = (const float*)d_in[3];
  const float* attnB  = (const float*)d_in[4];
  const float* combW  = (const float*)d_in[5];
  const float* combB  = (const float*)d_in[6];
  const float* Wih    = (const float*)d_in[7];
  const float* Whh    = (const float*)d_in[8];
  const float* bih    = (const float*)d_in[9];
  const float* bhh    = (const float*)d_in[10];
  const float* o1W    = (const float*)d_in[11];
  const float* o1b    = (const float*)d_in[12];
  const float* o2W    = (const float*)d_in[13];
  const float* o2b    = (const float*)d_in[14];

  float* out_y   = (float*)d_out;                          // (B, T, D)
  float* out_h   = out_y + (size_t)B_ * TOUT_ * D_;        // (L, B, H)
  float* out_att = out_h + (size_t)L_ * B_ * H_;           // (B, T, S)

  // ---- workspace carve (all offsets 256B-aligned) ----
  char* ws = (char*)d_ws;
  size_t off = 0;
  auto carve = [&](size_t bytes) { void* p = ws + off; off += (bytes + 255) & ~(size_t)255; return p; };
  float*  h_ws   = (float*)carve((size_t)L_ * B_ * H_ * 4);       // 2 MB
  __bf16* cat    = (__bf16*)carve((size_t)B_ * 2048 * 2);         // 1 MB
  __bf16* h0b    = (__bf16*)carve((size_t)B_ * H_ * 2);
  __bf16* h1b    = (__bf16*)carve((size_t)B_ * H_ * 2);
  __bf16* inpb   = (__bf16*)carve((size_t)B_ * H_ * 2);
  __bf16* xcb    = (__bf16*)carve((size_t)B_ * D_ * 2);
  __bf16* y1b    = (__bf16*)carve((size_t)B_ * D_ * 2);
  float*  logits = (float*)carve((size_t)B_ * SIN_ * 4);
  float*  aw     = (float*)carve((size_t)B_ * SIN_ * 4);
  float*  gi     = (float*)carve((size_t)B_ * 3 * H_ * 4);        // 3 MB
  float*  gh     = (float*)carve((size_t)B_ * 3 * H_ * 4);        // 3 MB
  __bf16* attnWb = (__bf16*)carve((size_t)SIN_ * 2048 * 2);
  __bf16* combWb = (__bf16*)carve((size_t)D_ * 2048 * 2);
  __bf16* WihB   = (__bf16*)carve((size_t)L_ * 3 * H_ * D_ * 2);  // 12 MB
  __bf16* WhhB   = (__bf16*)carve((size_t)L_ * 3 * H_ * H_ * 2);  // 12 MB
  __bf16* o1Wb   = (__bf16*)carve((size_t)D_ * H_ * 2);
  __bf16* o2Wb   = (__bf16*)carve((size_t)D_ * D_ * 2);
  (void)ws_size; (void)in_sizes; (void)n_in; (void)out_size;

  auto cdiv = [](long n, long b) { return (int)((n + b - 1) / b); };

  // ---- once per launch: convert weights to bf16, copy hidden state ----
  k_f2bf<<<cdiv((long)SIN_ * 2048, 256), 256, 0, stream>>>(attnW, attnWb, SIN_ * 2048);
  k_f2bf<<<cdiv((long)D_ * 2048, 256), 256, 0, stream>>>(combW, combWb, D_ * 2048);
  k_f2bf<<<cdiv((long)L_ * 3 * H_ * D_, 256), 256, 0, stream>>>(Wih, WihB, L_ * 3 * H_ * D_);
  k_f2bf<<<cdiv((long)L_ * 3 * H_ * H_, 256), 256, 0, stream>>>(Whh, WhhB, L_ * 3 * H_ * H_);
  k_f2bf<<<cdiv((long)D_ * H_, 256), 256, 0, stream>>>(o1W, o1Wb, D_ * H_);
  k_f2bf<<<cdiv((long)D_ * D_, 256), 256, 0, stream>>>(o2W, o2Wb, D_ * D_);
  k_copy_f32<<<cdiv((long)L_ * B_ * H_, 256), 256, 0, stream>>>(hidden, h_ws, L_ * B_ * H_);

  const int ewBH = cdiv((long)B_ * H_, 256);

  for (int t = 0; t < TOUT_; ++t) {
    // cat = [x_t, h1] (bf16), snapshot h0/h1 as bf16
    k_pack_step<<<ewBH, 256, 0, stream>>>(target, t, h_ws, cat, h0b, h1b);
    // attention logits (256x96) = cat(256x2048) @ attnW^T  (small: favor parallelism)
    launch_gemm<1, 2, false, false>(stream, cat, attnWb, attnB, logits, nullptr, B_, SIN_, 2048, SIN_);
    // softmax -> aw (+ write attn_weights output)
    k_softmax96<<<B_ / 8, 256, 0, stream>>>(logits, aw, out_att, t);
    // w_enc -> second half of cat (bf16)
    k_wenc<<<ewBH, 256, 0, stream>>>(aw, enc, cat);
    // xc = relu(cat @ combW^T + b) -> bf16
    launch_gemm<2, 4, true, true>(stream, cat, combWb, combB, nullptr, xcb, B_, D_, 2048, D_);
    // GRU layer 0
    launch_gemm<2, 4, false, false>(stream, xcb, WihB, bih, gi, nullptr, B_, 3 * H_, D_, 3 * H_);
    launch_gemm<2, 4, false, false>(stream, h0b, WhhB, bhh, gh, nullptr, B_, 3 * H_, H_, 3 * H_);
    k_gru_gate<<<ewBH, 256, 0, stream>>>(gi, gh, h_ws, inpb);
    // GRU layer 1
    launch_gemm<2, 4, false, false>(stream, inpb, WihB + (size_t)3 * H_ * D_, bih + 3 * H_,
                                    gi, nullptr, B_, 3 * H_, H_, 3 * H_);
    launch_gemm<2, 4, false, false>(stream, h1b, WhhB + (size_t)3 * H_ * H_, bhh + 3 * H_,
                                    gh, nullptr, B_, 3 * H_, H_, 3 * H_);
    k_gru_gate<<<ewBH, 256, 0, stream>>>(gi, gh, h_ws + (size_t)B_ * H_, inpb);
    // output head: y = (h1' @ o1W^T + b1) @ o2W^T + b2, strided into (B,T,D)
    launch_gemm<2, 4, false, true>(stream, inpb, o1Wb, o1b, nullptr, y1b, B_, D_, H_, D_);
    launch_gemm<2, 4, false, false>(stream, y1b, o2Wb, o2b, out_y + (size_t)t * D_, nullptr,
                                    B_, D_, D_, TOUT_ * D_);
  }

  // final hidden state
  k_copy_f32<<<cdiv((long)L_ * B_ * H_, 256), 256, 0, stream>>>(h_ws, out_h, L_ * B_ * H_);
}
